// Attention_21517786152986
// MI455X (gfx1250) — matrix operand order repaired and checked by reading, not compiled, not run on hardware
//
#include <hip/hip_runtime.h>
#include <cstdint>
#include <cstddef>

// ---------------------------------------------------------------------------
// Types
// ---------------------------------------------------------------------------
typedef __bf16 bf16_t;
typedef __attribute__((ext_vector_type(8)))  float  v8f;
typedef __attribute__((ext_vector_type(8)))  bf16_t v8bf;
typedef __attribute__((ext_vector_type(16))) bf16_t v16bf;

#define B_   2
#define N_   2048
#define H_   16
#define DH_  64
#define DIM_ 1024
#define M_   (B_ * N_)     // 4096 tokens
#define LDT  40            // padded LDS row stride (32 + 8 bf16) -> conflict-free b128

// ---------------------------------------------------------------------------
// WMMA helpers (CDNA5: V_WMMA_F32_16X16X32_BF16, wave32)
// ---------------------------------------------------------------------------
__device__ __forceinline__ v8f wmma_bf16(v16bf a, v16bf b, v8f c) {
  // (neg_a, A, neg_b, B, c_mod, C, reuse_a, reuse_b)
  return __builtin_amdgcn_wmma_f32_16x16x32_bf16(false, a, false, b, (short)0, c,
                                                 false, false);
}

__device__ __forceinline__ v8f zero_v8f() {
  v8f z;
#pragma unroll
  for (int i = 0; i < 8; ++i) z[i] = 0.0f;
  return z;
}

// A operand (16x32, M x K): lane = M row (L%16); elements hold K = 16*half + 0..15
// (same k order as load_fragB).
__device__ __forceinline__ v16bf load_fragA(const bf16_t* base, int row0, int stride,
                                            int coff, int lr, int half) {
  const bf16_t* p = base + (size_t)(row0 + lr) * stride + coff;
  v8bf lo = *(const v8bf*)(p + 16 * half);
  v8bf hi = *(const v8bf*)(p + 16 * half + 8);
  v16bf r;
#pragma unroll
  for (int i = 0; i < 8; ++i) { r[i] = lo[i]; r[i + 8] = hi[i]; }
  return r;
}

// B operand (32x16, K x N) from a row-major [N][K] buffer (i.e. B^T):
// lane = N column (L%16); elements hold K = 16*half + 0..15 (same k order as load_fragA).
__device__ __forceinline__ v16bf load_fragB(const bf16_t* base, int row0, int stride,
                                            int coff, int lr, int half) {
  const bf16_t* p = base + (size_t)(row0 + lr) * stride + coff + 16 * half;
  v8bf lo = *(const v8bf*)(p);
  v8bf hi = *(const v8bf*)(p + 8);
  v16bf r;
#pragma unroll
  for (int i = 0; i < 8; ++i) { r[i] = lo[i]; r[i + 8] = hi[i]; }
  return r;
}

// ---------------------------------------------------------------------------
// Elementwise conversion kernels
// ---------------------------------------------------------------------------
__global__ __launch_bounds__(256) void attn_cvt_bf16(const float* __restrict__ in,
                                                     bf16_t* __restrict__ out, int n) {
  int i = blockIdx.x * 256 + threadIdx.x;
  if (i < n) out[i] = (bf16_t)in[i];
}

// W [dim][dim] fp32 row-major  ->  WT [dim][dim] bf16 row-major (transposed)
__global__ __launch_bounds__(256) void attn_transpose_bf16(const float* __restrict__ in,
                                                           bf16_t* __restrict__ out,
                                                           int dim) {
  __shared__ float tile[32][33];
  const int bx = blockIdx.x * 32, by = blockIdx.y * 32;
  const int tx = threadIdx.x, ty = threadIdx.y;  // 32 x 8
#pragma unroll
  for (int i = 0; i < 32; i += 8)
    tile[ty + i][tx] = in[(size_t)(by + ty + i) * dim + bx + tx];
  __syncthreads();
#pragma unroll
  for (int i = 0; i < 32; i += 8)
    out[(size_t)(bx + ty + i) * dim + by + tx] = (bf16_t)tile[tx][ty + i];
}

// ---------------------------------------------------------------------------
// Shared GEMM mainloop: C[128x128] += A[M,K] * B[K,N], B given transposed [N][K].
// 256 threads = 8 waves; wave tile 32(M) x 64(N) = 2x4 wmma tiles.
// ---------------------------------------------------------------------------
__device__ __forceinline__ void gemm_mainloop(const bf16_t* __restrict__ A,
                                              const bf16_t* __restrict__ BT,
                                              int m0, int n0, int Kdim,
                                              bf16_t* As, bf16_t* Bs, v8f acc[2][4]) {
  const int tid  = threadIdx.x;
  const int w    = tid >> 5, lane = tid & 31;
  const int lr   = lane & 15, half = lane >> 4;
  const int wm   = (w & 3) * 32;
  const int wn   = (w >> 2) * 64;

  for (int kt = 0; kt < Kdim; kt += 32) {
#pragma unroll
    for (int s = tid; s < 512; s += 256) {   // 128 rows x 32 cols, 8 bf16 per thread
      int r = s >> 2, c = (s & 3) * 8;
      *(v8bf*)&As[r * LDT + c] = *(const v8bf*)(A  + (size_t)(m0 + r) * Kdim + kt + c);
      *(v8bf*)&Bs[r * LDT + c] = *(const v8bf*)(BT + (size_t)(n0 + r) * Kdim + kt + c);
    }
    __syncthreads();

    v16bf af[2], bw[4];
    af[0] = load_fragA(As, wm,      LDT, 0, lr, half);
    af[1] = load_fragA(As, wm + 16, LDT, 0, lr, half);
#pragma unroll
    for (int j = 0; j < 4; ++j) bw[j] = load_fragB(Bs, wn + 16 * j, LDT, 0, lr, half);
#pragma unroll
    for (int i = 0; i < 2; ++i)
#pragma unroll
      for (int j = 0; j < 4; ++j)
        acc[i][j] = wmma_bf16(af[i], bw[j], acc[i][j]);
    __syncthreads();
  }
}

// QKV projection: out laid out [B,H,N,DH] bf16 (tok = m index, feat = n index)
__global__ __launch_bounds__(256) void attn_gemm_qkv(const bf16_t* __restrict__ A,
                                                     const bf16_t* __restrict__ BT,
                                                     bf16_t* __restrict__ out) {
  __shared__ bf16_t As[128 * LDT];
  __shared__ bf16_t Bs[128 * LDT];
  const int m0 = blockIdx.x * 128, n0 = blockIdx.y * 128;
  v8f acc[2][4];
#pragma unroll
  for (int i = 0; i < 2; ++i)
#pragma unroll
    for (int j = 0; j < 4; ++j) acc[i][j] = zero_v8f();

  gemm_mainloop(A, BT, m0, n0, DIM_, As, Bs, acc);

  const int lane = threadIdx.x & 31, w = threadIdx.x >> 5;
  const int lr = lane & 15, half = lane >> 4;
  const int wm = (w & 3) * 32, wn = (w >> 2) * 64;
#pragma unroll
  for (int i = 0; i < 2; ++i)
#pragma unroll
    for (int j = 0; j < 4; ++j) {
      const int feat = n0 + wn + 16 * j + lr;
      const int h = feat >> 6, d = feat & 63;
#pragma unroll
      for (int v = 0; v < 8; ++v) {
        const int tok = m0 + wm + 16 * i + v + 8 * half;
        const int b = tok >> 11, nn = tok & (N_ - 1);
        out[(((size_t)(b * H_ + h) * N_ + nn) << 6) + d] = (bf16_t)acc[i][j][v];
      }
    }
}

// Output projection: fp32 store + bias
__global__ __launch_bounds__(256) void attn_gemm_out(const bf16_t* __restrict__ A,
                                                     const bf16_t* __restrict__ BT,
                                                     const float* __restrict__ bias,
                                                     float* __restrict__ out) {
  __shared__ bf16_t As[128 * LDT];
  __shared__ bf16_t Bs[128 * LDT];
  const int m0 = blockIdx.x * 128, n0 = blockIdx.y * 128;
  v8f acc[2][4];
#pragma unroll
  for (int i = 0; i < 2; ++i)
#pragma unroll
    for (int j = 0; j < 4; ++j) acc[i][j] = zero_v8f();

  gemm_mainloop(A, BT, m0, n0, DIM_, As, Bs, acc);

  const int lane = threadIdx.x & 31, w = threadIdx.x >> 5;
  const int lr = lane & 15, half = lane >> 4;
  const int wm = (w & 3) * 32, wn = (w >> 2) * 64;
#pragma unroll
  for (int i = 0; i < 2; ++i)
#pragma unroll
    for (int j = 0; j < 4; ++j) {
      const int feat = n0 + wn + 16 * j + lr;
      const float bb = bias[feat];
#pragma unroll
      for (int v = 0; v < 8; ++v) {
        const int tok = m0 + wm + 16 * i + v + 8 * half;
        out[(size_t)tok * DIM_ + feat] = acc[i][j][v] + bb;
      }
    }
}

// ---------------------------------------------------------------------------
// RoPE (+ optional scale) in-place on [B,H,N,DH] bf16; angles = pos_emb[N][DH].
// rot_half: out[d] = t[d]*cos(a[d]) - t[d+32]*sin(a[d]);
//           out[d+32] = t[d+32]*cos(a[d+32]) + t[d]*sin(a[d+32])
// ---------------------------------------------------------------------------
__global__ __launch_bounds__(256) void attn_rope(bf16_t* __restrict__ T,
                                                 const float* __restrict__ pe,
                                                 float scale) {
  const int idx = blockIdx.x * 256 + threadIdx.x;   // B*H*N*32 threads
  const int d  = idx & 31;
  const int nn = (idx >> 5) & (N_ - 1);
  const int bh = idx >> 16;
  const size_t base = ((size_t)bh * N_ + nn) * DH_;
  const float a0 = pe[nn * DH_ + d];
  const float a1 = pe[nn * DH_ + d + 32];
  const float t0 = (float)T[base + d];
  const float t1 = (float)T[base + d + 32];
  const float o0 = t0 * __cosf(a0) - t1 * __sinf(a0);
  const float o1 = t1 * __cosf(a1) + t0 * __sinf(a1);
  T[base + d]      = (bf16_t)(o0 * scale);
  T[base + d + 32] = (bf16_t)(o1 * scale);
}

// ---------------------------------------------------------------------------
// Flash attention, causal. Grid = B*H*(N/64); 128 threads = 4 waves.
// Wave owns 16 Q rows; KV processed in chunks of 32 via LDS.
// S: 4x v_wmma, online fp32 softmax (16-lane shfl_xor reductions), O: 4x v_wmma.
// ---------------------------------------------------------------------------
__global__ __launch_bounds__(128) void attn_flash(const bf16_t* __restrict__ Q,
                                                  const bf16_t* __restrict__ K,
                                                  const bf16_t* __restrict__ V,
                                                  bf16_t* __restrict__ O) {
  __shared__ bf16_t Ks[32 * 72];      // K chunk  [kv 32][d 64], stride 72
  __shared__ bf16_t Vt[64 * 40];      // V^T chunk [d 64][kv 32], stride 40
  __shared__ bf16_t Pw[4][16 * 40];   // per-wave P [m 16][kv 32], stride 40

  const int blk = blockIdx.x;
  const int qb  = blk & 31;           // N/64 = 32 q-blocks
  const int bh  = blk >> 5;           // b*H + h
  const int m0  = qb * 64;

  const bf16_t* Qp = Q + (size_t)bh * N_ * DH_;
  const bf16_t* Kp = K + (size_t)bh * N_ * DH_;
  const bf16_t* Vp = V + (size_t)bh * N_ * DH_;

  const int tid = threadIdx.x;
  const int w = tid >> 5, lane = tid & 31;
  const int lr = lane & 15, half = lane >> 4;
  const int qrow0 = m0 + w * 16;

  // Q fragments (A operand), d-chunks 0 and 1, loaded once.
  const v16bf aq0 = load_fragA(Qp, qrow0, DH_, 0,  lr, half);
  const v16bf aq1 = load_fragA(Qp, qrow0, DH_, 32, lr, half);

  v8f acc[4];
#pragma unroll
  for (int t = 0; t < 4; ++t) acc[t] = zero_v8f();
  float rmax[8], rsum[8];
#pragma unroll
  for (int v = 0; v < 8; ++v) { rmax[v] = -1e30f; rsum[v] = 0.0f; }

  const int jend = m0 + 64;
  for (int j0 = 0; j0 < jend; j0 += 32) {
    // Stage K chunk (row-major) and V chunk (transposed) into LDS.
#pragma unroll
    for (int s = tid; s < 256; s += 128) {
      const int r = s >> 3, c = (s & 7) * 8;
      *(v8bf*)&Ks[r * 72 + c] = *(const v8bf*)(Kp + (size_t)(j0 + r) * DH_ + c);
      v8bf vv = *(const v8bf*)(Vp + (size_t)(j0 + r) * DH_ + c);
#pragma unroll
      for (int i = 0; i < 8; ++i) Vt[(c + i) * 40 + r] = vv[i];
    }
    if (j0 + 32 < jend) {  // gfx1250 global_prefetch of next KV chunk
      __builtin_prefetch((const void*)(Kp + (size_t)(j0 + 32) * DH_ + lane * 8), 0, 1);
      __builtin_prefetch((const void*)(Vp + (size_t)(j0 + 32) * DH_ + lane * 8), 0, 1);
    }
    __syncthreads();

    if (j0 <= qrow0 + 15) {   // wave-uniform causal skip (EXEC stays all-1 for WMMA)
      // S = Q * K^T  (two 16-wide n-tiles, d split into two K=32 chunks)
      const v16bf bk00 = load_fragB(Ks, 0,  72, 0,  lr, half);
      const v16bf bk01 = load_fragB(Ks, 0,  72, 32, lr, half);
      const v16bf bk10 = load_fragB(Ks, 16, 72, 0,  lr, half);
      const v16bf bk11 = load_fragB(Ks, 16, 72, 32, lr, half);
      v8f s0 = zero_v8f(), s1 = zero_v8f();
      s0 = wmma_bf16(aq0, bk00, s0);
      s0 = wmma_bf16(aq1, bk01, s0);
      s1 = wmma_bf16(aq0, bk10, s1);
      s1 = wmma_bf16(aq1, bk11, s1);

      const int kv0 = j0 + lr, kv1 = j0 + 16 + lr;
#pragma unroll
      for (int v = 0; v < 8; ++v) {
        const int mrow = qrow0 + v + 8 * half;
        float e0 = (kv0 <= mrow) ? s0[v] : -1e30f;
        float e1 = (kv1 <= mrow) ? s1[v] : -1e30f;
        float cm = fmaxf(e0, e1);
#pragma unroll
        for (int off = 8; off; off >>= 1) cm = fmaxf(cm, __shfl_xor(cm, off, 32));
        const float mn = fmaxf(rmax[v], cm);
        const float p0 = __expf(e0 - mn);
        const float p1 = __expf(e1 - mn);
        const float alpha = __expf(rmax[v] - mn);
        rmax[v] = mn;
        float ps = p0 + p1;
#pragma unroll
        for (int off = 8; off; off >>= 1) ps += __shfl_xor(ps, off, 32);
        rsum[v] = rsum[v] * alpha + ps;
#pragma unroll
        for (int t = 0; t < 4; ++t) acc[t][v] *= alpha;
        const int m = v + 8 * half;
        Pw[w][m * 40 + lr]      = (bf16_t)p0;
        Pw[w][m * 40 + 16 + lr] = (bf16_t)p1;
      }
      // Per-wave LDS RAW turn-around (belt & braces on top of in-order DS).
      asm volatile("s_wait_dscnt 0" ::: "memory");

      // O += P * V   (A = P 16x32, B = V 32x64 via Vt)
      const v16bf ap = load_fragA(&Pw[w][0], 0, 40, 0, lr, half);
#pragma unroll
      for (int t = 0; t < 4; ++t) {
        const v16bf bv = load_fragB(Vt, 16 * t, 40, 0, lr, half);
        acc[t] = wmma_bf16(ap, bv, acc[t]);
      }
    }
    __syncthreads();
  }

  // Normalize and store to [B, N, H*DH] bf16 for the output projection.
  const int b = bh >> 4, h = bh & 15;
#pragma unroll
  for (int v = 0; v < 8; ++v) {
    const float inv = 1.0f / rsum[v];
    const int nrow = qrow0 + v + 8 * half;
    const size_t rowoff = (((size_t)(b * N_ + nrow)) << 10) + ((size_t)h << 6);
#pragma unroll
    for (int t = 0; t < 4; ++t)
      O[rowoff + 16 * t + lr] = (bf16_t)(acc[t][v] * inv);
  }
}

// ---------------------------------------------------------------------------
// Host launcher
// ---------------------------------------------------------------------------
extern "C" void kernel_launch(void* const* d_in, const int* in_sizes, int n_in,
                              void* d_out, int out_size, void* d_ws, size_t ws_size,
                              hipStream_t stream) {
  (void)in_sizes; (void)n_in; (void)out_size; (void)ws_size;
  const float* x  = (const float*)d_in[0];
  const float* pe = (const float*)d_in[1];
  const float* Wq = (const float*)d_in[2];
  const float* Wk = (const float*)d_in[3];
  const float* Wv = (const float*)d_in[4];
  const float* Wo = (const float*)d_in[5];
  const float* bo = (const float*)d_in[6];
  float* out = (float*)d_out;

  size_t off = 0;
  auto take = [&](size_t bytes) -> void* {
    void* p = (char*)d_ws + off;
    off += (bytes + 255) & ~(size_t)255;
    return p;
  };
  bf16_t* xb  = (bf16_t*)take((size_t)M_ * DIM_ * 2);   // x in bf16
  bf16_t* WqT = (bf16_t*)take((size_t)DIM_ * DIM_ * 2); // W^T bf16
  bf16_t* WkT = (bf16_t*)take((size_t)DIM_ * DIM_ * 2);
  bf16_t* WvT = (bf16_t*)take((size_t)DIM_ * DIM_ * 2);
  bf16_t* WoT = (bf16_t*)take((size_t)DIM_ * DIM_ * 2);
  bf16_t* Qb  = (bf16_t*)take((size_t)M_ * DIM_ * 2);   // [B,H,N,DH]
  bf16_t* Kb  = (bf16_t*)take((size_t)M_ * DIM_ * 2);
  bf16_t* Vb  = (bf16_t*)take((size_t)M_ * DIM_ * 2);
  bf16_t* Ob  = (bf16_t*)take((size_t)M_ * DIM_ * 2);   // [B,N,H*DH]

  attn_cvt_bf16<<<(M_ * DIM_) / 256, 256, 0, stream>>>(x, xb, M_ * DIM_);

  dim3 tg(DIM_ / 32, DIM_ / 32), tb(32, 8);
  attn_transpose_bf16<<<tg, tb, 0, stream>>>(Wq, WqT, DIM_);
  attn_transpose_bf16<<<tg, tb, 0, stream>>>(Wk, WkT, DIM_);
  attn_transpose_bf16<<<tg, tb, 0, stream>>>(Wv, WvT, DIM_);
  attn_transpose_bf16<<<tg, tb, 0, stream>>>(Wo, WoT, DIM_);

  dim3 gg(M_ / 128, DIM_ / 128);
  attn_gemm_qkv<<<gg, 256, 0, stream>>>(xb, WqT, Qb);
  attn_gemm_qkv<<<gg, 256, 0, stream>>>(xb, WkT, Kb);
  attn_gemm_qkv<<<gg, 256, 0, stream>>>(xb, WvT, Vb);

  const int rope_threads = B_ * H_ * N_ * 32;  // one thread per (b,h,n,d<32) pair
  attn_rope<<<rope_threads / 256, 256, 0, stream>>>(Qb, pe, 0.125f);  // q * DH^-0.5
  attn_rope<<<rope_threads / 256, 256, 0, stream>>>(Kb, pe, 1.0f);

  attn_flash<<<B_ * H_ * (N_ / 64), 128, 0, stream>>>(Qb, Kb, Vb, Ob);

  attn_gemm_out<<<gg, 256, 0, stream>>>(Ob, WoT, bo, out);
}
